// HGCNEncoder_60000693125356
// MI455X (gfx1250) — matrix operand order repaired and checked by reading, not compiled, not run on hardware
//
#include <hip/hip_runtime.h>
#include <hip/hip_bf16.h>
#include <stdint.h>
#include <stddef.h>

// ---------------------------------------------------------------------------
// HGCN encoder for MI455X (gfx1250): bf16 WMMA GEMMs + atomic edge aggregation
// GEMM tiles are 64x128 (BN widened: weights are L2-resident, activations are
// the HBM-bound operand, so fewer column blocks => fewer A re-reads).
// ---------------------------------------------------------------------------

typedef __attribute__((ext_vector_type(16))) __bf16 v16bf;
typedef __attribute__((ext_vector_type(8)))  float  v8f;

#define NN 100000   // total nodes
#define EE 800000   // edges

// round-to-nearest-even f32 -> bf16, pack two into a dword (lo = even K)
static __device__ __forceinline__ uint32_t pack_bf16(float lo, float hi) {
  union { float f; uint32_t u; } a, b;
  a.f = lo; b.f = hi;
  uint32_t ua = (a.u + 0x7FFFu + ((a.u >> 16) & 1u)) >> 16;
  uint32_t ub = (b.u + 0x7FFFu + ((b.u >> 16) & 1u)) >> 16;
  return (ua & 0xFFFFu) | (ub << 16);
}

// ---------------------------------------------------------------------------
// Generic GEMM: C[M,N] = act(A[M,K]) @ B[K,N] (+ bias), f32 in, bf16 WMMA,
// f32 accumulate. Block = 128 threads (4 waves), tile = 64x128, K step 32.
// N must be a multiple of 128, K a multiple of 32 (true for all five GEMMs:
// N in {256,512,128}, K in {128,256,64,512}); M is guarded.
// Per wave: 16 rows x 128 cols = 8 WMMA tiles per K step.
// ---------------------------------------------------------------------------
template<bool RELU_A, bool HAS_BIAS>
__global__ __launch_bounds__(128)
void gemm_wmma_bf16(const float* __restrict__ A, const float* __restrict__ B,
                    const float* __restrict__ bias, float* __restrict__ C,
                    int M, int N, int K) {
  // K-pair packed tiles (uint = two bf16 along K). Padded against bank conflicts.
  __shared__ uint32_t As2[64][18];    // [m][k/2]
  __shared__ uint32_t Bs2[16][132];   // [k/2][n]

  const int tid  = threadIdx.x;
  const int lane = tid & 31;
  const int wave = tid >> 5;                 // 0..3 -> row sub-tile
  const int rowBase = blockIdx.y * 64;
  const int colBase = blockIdx.x * 128;

  const v8f zero = {0.f,0.f,0.f,0.f,0.f,0.f,0.f,0.f};
  v8f acc[8] = {zero, zero, zero, zero, zero, zero, zero, zero};

  // ISA fragment coordinates (wave32, 16x16x32 bf16)
  const int mfrag = (lane & 15) + wave * 16; // A row in LDS tile
  const int ksel  = lane >> 4;               // lane half selects K phase
  const int kpA   = ksel * 4;                // A: kbase/2 (kbase = 0 or 8)
  const int nfrag = lane & 15;               // B/C column
  const int kpB   = ksel * 4;                // B: kbase/2 (kbase = 0 or 8)

  for (int k0 = 0; k0 < K; k0 += 32) {
    // ---- stage A tile: 64 rows x 16 k-pairs, convert f32->bf16 (+ReLU) ----
    for (int i = tid; i < 64 * 16; i += 128) {
      const int r  = i >> 4;
      const int cp = i & 15;
      const int gr = rowBase + r;
      float f0 = 0.f, f1 = 0.f;
      if (gr < M) {
        const float* p = A + (size_t)gr * K + k0 + cp * 2;
        f0 = p[0]; f1 = p[1];
        if (RELU_A) { f0 = fmaxf(f0, 0.f); f1 = fmaxf(f1, 0.f); }
        if (k0 + 32 < K) __builtin_prefetch(p + 32, 0, 3);  // next K tile
      }
      As2[r][cp] = pack_bf16(f0, f1);
    }
    // ---- stage B tile: 16 k-pairs x 128 cols (weights: L2-resident) ----
    for (int i = tid; i < 16 * 128; i += 128) {
      const int rp = i >> 7;
      const int c  = i & 127;
      const float* p = B + (size_t)(k0 + rp * 2) * N + colBase + c;
      Bs2[rp][c] = pack_bf16(p[0], p[N]);
    }
    __syncthreads();

    // A fragment for this wave (16 bf16 per lane, ISA 16x32 layout)
    union { v16bf v; uint32_t u[8]; } af;
    #pragma unroll
    for (int j = 0; j < 4; ++j) af.u[j]     = As2[mfrag][kpA + j];      // K 0..7 (+8)
    #pragma unroll
    for (int j = 0; j < 4; ++j) af.u[4 + j] = As2[mfrag][kpA + 8 + j];  // K 16..23 (+8)

    #pragma unroll
    for (int t = 0; t < 8; ++t) {
      union { v16bf v; uint32_t u[8]; } bf;
      #pragma unroll
      for (int j = 0; j < 8; ++j) bf.u[j] = Bs2[kpB + j + (j & 4)][t * 16 + nfrag];
      acc[t] = __builtin_amdgcn_wmma_f32_16x16x32_bf16(
          false, af.v, false, bf.v, (short)0, acc[t], false, false);
    }
    __syncthreads();
  }

  // ---- store C (D layout: M = v + 8*(lane>=16), N = lane&15) ----
  #pragma unroll
  for (int t = 0; t < 8; ++t) {
    const int c = colBase + t * 16 + nfrag;
    float bv = 0.f;
    if (HAS_BIAS) bv = bias[c];
    #pragma unroll
    for (int v = 0; v < 8; ++v) {
      const int r = rowBase + wave * 16 + ksel * 8 + v;
      if (r < M) C[(size_t)r * N + c] = acc[t][v] + bv;
    }
  }
}

// ---------------------------------------------------------------------------
// Degree / normalization
// ---------------------------------------------------------------------------
__global__ void deg_init_kernel(float* __restrict__ deg) {
  int i = blockIdx.x * blockDim.x + threadIdx.x;
  if (i < NN) deg[i] = 1.0f;
}
__global__ void deg_count_kernel(const int* __restrict__ dst, float* __restrict__ deg) {
  int e = blockIdx.x * blockDim.x + threadIdx.x;
  if (e < EE) atomicAdd(&deg[dst[e]], 1.0f);
}
__global__ void dis_kernel(const float* __restrict__ deg, float* __restrict__ dis) {
  int i = blockIdx.x * blockDim.x + threadIdx.x;
  if (i < NN) dis[i] = rsqrtf(deg[i]);
}

// ---------------------------------------------------------------------------
// agg[j] = h[j] * dis[j]^2 + bias   (self-loop term + bias, fully overwrites)
// ---------------------------------------------------------------------------
template<int D, int BLK>
__global__ __launch_bounds__(BLK)
void agg_init_kernel(float* __restrict__ agg, const float* __restrict__ h,
                     const float* __restrict__ dis, const float* __restrict__ bias) {
  const int node = blockIdx.x;
  float s = dis[node]; s *= s;
  const float4* h4 = (const float4*)(h + (size_t)node * D);
  const float4* b4 = (const float4*)bias;
  float4*       a4 = (float4*)(agg + (size_t)node * D);
  for (int c = threadIdx.x; c < D / 4; c += BLK) {
    float4 v = h4[c];
    float4 b = b4[c];
    a4[c] = make_float4(v.x * s + b.x, v.y * s + b.y, v.z * s + b.z, v.w * s + b.w);
  }
}

// ---------------------------------------------------------------------------
// agg[dst] += h[src] * dis[src]*dis[dst]   (f32 global atomics, coalesced)
// D/4 threads cooperate per edge; 128-thread blocks
// ---------------------------------------------------------------------------
template<int D>
__global__ __launch_bounds__(128)
void agg_edge_kernel(float* __restrict__ agg, const float* __restrict__ h,
                     const float* __restrict__ dis,
                     const int* __restrict__ src, const int* __restrict__ dst) {
  constexpr int TPE = D / 4;          // threads per edge
  constexpr int EPB = 128 / TPE;      // edges per block
  const int e = blockIdx.x * EPB + threadIdx.x / TPE;
  if (e >= EE) return;
  const int l = threadIdx.x % TPE;
  const int s = src[e];
  const int d = dst[e];
  const float coef = dis[s] * dis[d];
  float4 v = ((const float4*)(h + (size_t)s * D))[l];
  float* ap = agg + (size_t)d * D + l * 4;
  atomicAdd(ap + 0, v.x * coef);
  atomicAdd(ap + 1, v.y * coef);
  atomicAdd(ap + 2, v.z * coef);
  atomicAdd(ap + 3, v.w * coef);
}

// ---------------------------------------------------------------------------
// launch
// ---------------------------------------------------------------------------
extern "C" void kernel_launch(void* const* d_in, const int* in_sizes, int n_in,
                              void* d_out, int out_size, void* d_ws, size_t ws_size,
                              hipStream_t stream) {
  // inputs in setup_inputs() order
  const float* x1  = (const float*)d_in[0];
  const float* x2  = (const float*)d_in[1];
  const float* x3  = (const float*)d_in[2];
  const float* Wp1 = (const float*)d_in[3];
  const float* bp1 = (const float*)d_in[4];
  const float* Wp2 = (const float*)d_in[5];
  const float* bp2 = (const float*)d_in[6];
  const float* Wp3 = (const float*)d_in[7];
  const float* bp3 = (const float*)d_in[8];
  const float* W1  = (const float*)d_in[9];
  const float* b1  = (const float*)d_in[10];
  const float* W2  = (const float*)d_in[11];
  const float* b2  = (const float*)d_in[12];
  const int*   ei  = (const int*)d_in[13];   // [2, E] flat
  const int* src = ei;
  const int* dst = ei + EE;
  float* out = (float*)d_out;

  // workspace carve-out
  size_t off = 0;
  auto carve = [&](size_t bytes) -> char* {
    char* p = (char*)d_ws + off;
    off += (bytes + 255) & ~(size_t)255;
    return p;
  };
  float* deg  = (float*)carve((size_t)NN * 4);
  float* dis  = (float*)carve((size_t)NN * 4);
  float* x    = (float*)carve((size_t)NN * 256 * 4);  // projected features
  float* h1   = (float*)carve((size_t)NN * 512 * 4);  // x @ W1
  float* agg1 = (float*)carve((size_t)NN * 512 * 4);  // layer-1 aggregation
  float* h2   = x;                                     // reuse: x dead after GEMM1

  // 1) degree + normalization
  deg_init_kernel<<<(NN + 255) / 256, 256, 0, stream>>>(deg);
  deg_count_kernel<<<(EE + 255) / 256, 256, 0, stream>>>(dst, deg);
  dis_kernel<<<(NN + 255) / 256, 256, 0, stream>>>(deg, dis);

  // 2) per-type projections -> x (f32, with bias); tile = 64 x 128
  gemm_wmma_bf16<false, true><<<dim3(256 / 128, (40000 + 63) / 64), 128, 0, stream>>>(
      x1, Wp1, bp1, x,                 40000, 256, 128);
  gemm_wmma_bf16<false, true><<<dim3(256 / 128, (30000 + 63) / 64), 128, 0, stream>>>(
      x2, Wp2, bp2, x + (size_t)40000 * 256, 30000, 256, 256);
  gemm_wmma_bf16<false, true><<<dim3(256 / 128, (30000 + 63) / 64), 128, 0, stream>>>(
      x3, Wp3, bp3, x + (size_t)70000 * 256, 30000, 256, 64);

  // 3) layer 1: h1 = x @ W1 (no bias; bias fused into agg init)
  gemm_wmma_bf16<false, false><<<dim3(512 / 128, (NN + 63) / 64), 128, 0, stream>>>(
      x, W1, nullptr, h1, NN, 512, 256);

  // 4) layer-1 aggregation: agg1 = scatter(h1) + h1*dis^2 + b1
  agg_init_kernel<512, 128><<<NN, 128, 0, stream>>>(agg1, h1, dis, b1);
  agg_edge_kernel<512><<<EE, 128, 0, stream>>>(agg1, h1, dis, src, dst);

  // 5) layer 2: h2 = relu(agg1) @ W2 (ReLU fused into A loader)
  gemm_wmma_bf16<true, false><<<dim3(128 / 128, (NN + 63) / 64), 128, 0, stream>>>(
      agg1, W2, nullptr, h2, NN, 128, 512);

  // 6) layer-2 aggregation straight into d_out (init fully overwrites poison)
  agg_init_kernel<128, 32><<<NN, 32, 0, stream>>>(out, h2, dis, b2);
  agg_edge_kernel<128><<<(EE + 3) / 4, 128, 0, stream>>>(out, h2, dis, src, dst);
}
